// Attention_7378753814681
// MI455X (gfx1250) — matrix operand order repaired and checked by reading, not compiled, not run on hardware
//
#include <hip/hip_runtime.h>

typedef __attribute__((ext_vector_type(16))) _Float16 v16h;
typedef __attribute__((ext_vector_type(8)))  _Float16 v8h;
typedef __attribute__((ext_vector_type(8)))  float    v8f;

#define NTOK  512
#define CDIM  512
#define NHEAD 8
#define HD    64
#define BATCH 32
#define QK_SCALE 0.125f   // 64^-0.5

static __device__ __forceinline__ v8f wmma_f16(v16h a, v16h b, v8f c) {
  // D = A(16x32 f16) * B(32x16 f16) + C(16x16 f32)
  return __builtin_amdgcn_wmma_f32_16x16x32_f16(
      /*neg_a=*/false, a, /*neg_b=*/false, b,
      /*c_mod=*/(short)0, c, /*reuse_a=*/false, /*reuse_b=*/false);
}

// Build A-fragment (16x32, f16) from two contiguous 8-half blocks.
// Element i holds K = (i&7) + 8*hi + (i>=8 ? 16 : 0).
static __device__ __forceinline__ v16h make_a(v8h lo, v8h hi8) {
  v16h a;
#pragma unroll
  for (int i = 0; i < 8; ++i) { a[i] = lo[i]; a[i + 8] = hi8[i]; }
  return a;
}

// ---------------------------------------------------------------------------
// Kernel 0: transpose + downconvert weights to f16 so B-fragments are
// contiguous per lane (B-frag: lane=n, element i -> K = (i&7) + 8*hi + (i>=8 ? 16 : 0)).
// wqkvt[n][k] = w_qkv[k][n]  (n in [0,1536), k in [0,512))
// wpt  [n][k] = w_proj[k][n] (n,k in [0,512))
// ---------------------------------------------------------------------------
__global__ void prep_weights(const float* __restrict__ w_qkv,
                             const float* __restrict__ w_proj,
                             _Float16* __restrict__ wqkvt,
                             _Float16* __restrict__ wpt) {
  const int t = blockIdx.x * blockDim.x + threadIdx.x;
  const int n1 = 3 * CDIM * CDIM;            // 786432
  if (t < n1) {
    const int n = t / CDIM;
    const int k = t - n * CDIM;
    wqkvt[t] = (_Float16)w_qkv[(size_t)k * (3 * CDIM) + n];
  }
  const int t2 = t - n1;
  if (t2 >= 0 && t2 < CDIM * CDIM) {
    const int n = t2 / CDIM;
    const int k = t2 - n * CDIM;
    wpt[t2] = (_Float16)w_proj[(size_t)k * CDIM + n];
  }
}

// ---------------------------------------------------------------------------
// Kernel 1: QKV GEMM.  One wave computes a 16(M) x 64(Nout) tile of
// [B*N,1536] = x @ w_qkv, splitting the result into:
//   Qh[bh][tok][d] (f16, pre-scaled by 1/8), Kh[bh][tok][d], Vt[bh][d][tok]
// ---------------------------------------------------------------------------
__global__ void __launch_bounds__(32) qkv_gemm(
    const float* __restrict__ x,          // [B*N][512] fp32
    const _Float16* __restrict__ wqkvt,   // [1536][512] f16 (n-major)
    _Float16* __restrict__ Qh,
    _Float16* __restrict__ Kh,
    _Float16* __restrict__ Vt) {
  const int lane = threadIdx.x;
  const int r  = lane & 15;
  const int hi = lane >> 4;
  const int nt = blockIdx.x % 24;         // 24 tiles of 64 over 1536
  const int mt = blockIdx.x / 24;         // 1024 tiles of 16 over 16384

  const float* xr = x + (size_t)(mt * 16 + r) * CDIM;
  const _Float16* wb[4];
#pragma unroll
  for (int jt = 0; jt < 4; ++jt)
    wb[jt] = wqkvt + (size_t)(nt * 64 + jt * 16 + r) * CDIM;

  v8f acc[4] = {v8f{}, v8f{}, v8f{}, v8f{}};
  for (int kc = 0; kc < 16; ++kc) {
    const int k0 = kc * 32;
    const float4 f0 = *(const float4*)(xr + k0 + 8 * hi);
    const float4 f1 = *(const float4*)(xr + k0 + 8 * hi + 4);
    const float4 f2 = *(const float4*)(xr + k0 + 16 + 8 * hi);
    const float4 f3 = *(const float4*)(xr + k0 + 16 + 8 * hi + 4);
    v16h a;
    a[0]=(_Float16)f0.x; a[1]=(_Float16)f0.y; a[2]=(_Float16)f0.z; a[3]=(_Float16)f0.w;
    a[4]=(_Float16)f1.x; a[5]=(_Float16)f1.y; a[6]=(_Float16)f1.z; a[7]=(_Float16)f1.w;
    a[8]=(_Float16)f2.x; a[9]=(_Float16)f2.y; a[10]=(_Float16)f2.z; a[11]=(_Float16)f2.w;
    a[12]=(_Float16)f3.x; a[13]=(_Float16)f3.y; a[14]=(_Float16)f3.z; a[15]=(_Float16)f3.w;
#pragma unroll
    for (int jt = 0; jt < 4; ++jt) {
      const v16h b = make_a(*(const v8h*)(wb[jt] + k0 + 8 * hi), *(const v8h*)(wb[jt] + k0 + 16 + 8 * hi));
      acc[jt] = wmma_f16(a, b, acc[jt]);
    }
  }

#pragma unroll
  for (int jt = 0; jt < 4; ++jt) {
    const int g   = nt * 64 + jt * 16 + r;  // global out column (tile never straddles 512)
    const int sec = g >> 9;                 // 0=Q 1=K 2=V
    const int c   = g & 511;
    const int h   = c >> 6;
    const int d   = c & 63;
#pragma unroll
    for (int v = 0; v < 8; ++v) {
      const int m    = mt * 16 + v + 8 * hi;   // row in [0, B*N)
      const int bidx = m >> 9;
      const int tok  = m & 511;
      const int bh   = bidx * NHEAD + h;
      const float val = acc[jt][v];
      if (sec == 0)      Qh[((size_t)bh * NTOK + tok) * HD + d] = (_Float16)(val * QK_SCALE);
      else if (sec == 1) Kh[((size_t)bh * NTOK + tok) * HD + d] = (_Float16)val;
      else               Vt[((size_t)bh * HD + d) * NTOK + tok] = (_Float16)val;
    }
  }
}

// ---------------------------------------------------------------------------
// Kernel 2: attention. One wave per (b, h, 16-row strip):
//   S = Qs @ K^T + mask  (32 col tiles, 2 WMMAs each)   -> LDS f32
//   rowwise max / exp / sum (lanes 0..15)               -> P f16 in LDS
//   out = (P @ V) * (1/sum)                              -> attn16[B*N][512]
// ---------------------------------------------------------------------------
__global__ void __launch_bounds__(32) attn_kernel(
    const _Float16* __restrict__ Qh,
    const _Float16* __restrict__ Kh,
    const _Float16* __restrict__ Vt,
    const float* __restrict__ mask,     // [512][512] fp32
    _Float16* __restrict__ attn16) {    // [B*N][512] f16
  __shared__ __align__(16) float    Sbuf[16][NTOK];   // 32 KB
  __shared__ __align__(16) _Float16 Pbuf[16][NTOK];   // 16 KB
  __shared__ float invs[16];

  const int lane = threadIdx.x;
  const int r  = lane & 15;
  const int hi = lane >> 4;
  const int bh   = blockIdx.x >> 5;    // 0..255  (b*8 + h)
  const int rt   = blockIdx.x & 31;
  const int row0 = rt * 16;

  // Q A-fragments (kept in registers across all 32 column tiles)
  const _Float16* qb = Qh + ((size_t)bh * NTOK + row0 + r) * HD;
  const v16h aq0 = make_a(*(const v8h*)(qb +      8 * hi), *(const v8h*)(qb + 16 + 8 * hi));
  const v16h aq1 = make_a(*(const v8h*)(qb + 32 + 8 * hi), *(const v8h*)(qb + 48 + 8 * hi));

  for (int j = 0; j < 32; ++j) {
    v8f acc = {};
    const _Float16* kb = Kh + ((size_t)bh * NTOK + j * 16 + r) * HD;
    const v16h b0 = make_a(*(const v8h*)(kb +      8 * hi), *(const v8h*)(kb + 16 + 8 * hi));
    const v16h b1 = make_a(*(const v8h*)(kb + 32 + 8 * hi), *(const v8h*)(kb + 48 + 8 * hi));
    acc = wmma_f16(aq0, b0, acc);
    acc = wmma_f16(aq1, b1, acc);
    const int cn = j * 16 + r;
#pragma unroll
    for (int v = 0; v < 8; ++v) {
      const int rr = v + 8 * hi;
      Sbuf[rr][cn] = acc[v] + mask[(size_t)(row0 + rr) * NTOK + cn];
    }
  }
  __syncthreads();

  if (lane < 16) {
    const float4* srow = (const float4*)Sbuf[lane];
    float mx = -3.0e38f;
    for (int c4 = 0; c4 < NTOK / 4; ++c4) {
      const float4 sv = srow[c4];
      mx = fmaxf(mx, fmaxf(fmaxf(sv.x, sv.y), fmaxf(sv.z, sv.w)));
    }
    float sum = 0.f;
    for (int c = 0; c < NTOK; ++c) {
      const float e = __expf(Sbuf[lane][c] - mx);
      sum += e;
      Pbuf[lane][c] = (_Float16)e;   // unnormalized; 1/sum folded into epilogue
    }
    invs[lane] = 1.0f / sum;
  }
  __syncthreads();

  const int b = bh >> 3;
  const int h = bh & 7;
#pragma unroll
  for (int t = 0; t < 4; ++t) {
    v8f acc = {};
    const _Float16* vb = Vt + ((size_t)bh * HD + t * 16 + r) * NTOK;
    for (int kc = 0; kc < 16; ++kc) {
      const v16h a = make_a(*(const v8h*)(&Pbuf[r][kc * 32 + 8 * hi]),
                            *(const v8h*)(&Pbuf[r][kc * 32 + 16 + 8 * hi]));
      const v16h bf = make_a(*(const v8h*)(vb + kc * 32 + 8 * hi), *(const v8h*)(vb + kc * 32 + 16 + 8 * hi));
      acc = wmma_f16(a, bf, acc);
    }
#pragma unroll
    for (int v = 0; v < 8; ++v) {
      const int rr  = v + 8 * hi;
      const int tok = row0 + rr;
      attn16[((size_t)b * NTOK + tok) * CDIM + h * HD + t * 16 + r] =
          (_Float16)(acc[v] * invs[rr]);
    }
  }
}

// ---------------------------------------------------------------------------
// Kernel 3: output projection. One wave per 16x64 tile of [16384][512]:
//   out = attn16 @ w_proj + b_proj   (fp32 out)
// ---------------------------------------------------------------------------
__global__ void __launch_bounds__(32) proj_gemm(
    const _Float16* __restrict__ attn16,  // [B*N][512] f16
    const _Float16* __restrict__ wpt,     // [512][512] f16 (n-major)
    const float* __restrict__ b_proj,     // [512]
    float* __restrict__ out) {            // [B*N][512] fp32
  const int lane = threadIdx.x;
  const int r  = lane & 15;
  const int hi = lane >> 4;
  const int nt = blockIdx.x & 7;          // 8 tiles of 64 over 512
  const int mt = blockIdx.x >> 3;         // 1024 tiles of 16 over 16384

  const _Float16* ar = attn16 + (size_t)(mt * 16 + r) * CDIM;
  const _Float16* wb[4];
#pragma unroll
  for (int jt = 0; jt < 4; ++jt)
    wb[jt] = wpt + (size_t)(nt * 64 + jt * 16 + r) * CDIM;

  v8f acc[4] = {v8f{}, v8f{}, v8f{}, v8f{}};
  for (int kc = 0; kc < 16; ++kc) {
    const int k0 = kc * 32;
    const v16h a = make_a(*(const v8h*)(ar + k0 + 8 * hi),
                          *(const v8h*)(ar + k0 + 16 + 8 * hi));
#pragma unroll
    for (int jt = 0; jt < 4; ++jt) {
      const v16h b = make_a(*(const v8h*)(wb[jt] + k0 + 8 * hi), *(const v8h*)(wb[jt] + k0 + 16 + 8 * hi));
      acc[jt] = wmma_f16(a, b, acc[jt]);
    }
  }

#pragma unroll
  for (int jt = 0; jt < 4; ++jt) {
    const int col = nt * 64 + jt * 16 + r;
    const float bias = b_proj[col];
#pragma unroll
    for (int v = 0; v < 8; ++v) {
      const int m = mt * 16 + v + 8 * hi;
      out[(size_t)m * CDIM + col] = acc[jt][v] + bias;
    }
  }
}

// ---------------------------------------------------------------------------
extern "C" void kernel_launch(void* const* d_in, const int* in_sizes, int n_in,
                              void* d_out, int out_size, void* d_ws, size_t ws_size,
                              hipStream_t stream) {
  const float* x      = (const float*)d_in[0];  // [32,512,512]
  const float* w_qkv  = (const float*)d_in[1];  // [512,1536]
  const float* w_proj = (const float*)d_in[2];  // [512,512]
  const float* b_proj = (const float*)d_in[3];  // [512]
  const float* mask   = (const float*)d_in[4];  // [1,1,512,512]
  float* out = (float*)d_out;

  char* ws = (char*)d_ws;
  size_t off = 0;
  auto alloc = [&](size_t bytes) -> void* {
    void* p = ws + off;
    off += (bytes + 255) & ~(size_t)255;
    return p;
  };
  const size_t qkvh_elems = (size_t)BATCH * NHEAD * NTOK * HD;   // 8.4M
  _Float16* wqkvt  = (_Float16*)alloc((size_t)3 * CDIM * CDIM * 2);
  _Float16* wpt    = (_Float16*)alloc((size_t)CDIM * CDIM * 2);
  _Float16* Qh     = (_Float16*)alloc(qkvh_elems * 2);
  _Float16* Kh     = (_Float16*)alloc(qkvh_elems * 2);
  _Float16* Vt     = (_Float16*)alloc(qkvh_elems * 2);
  _Float16* attn16 = (_Float16*)alloc((size_t)BATCH * NTOK * CDIM * 2);
  if (off > ws_size) return;  // insufficient scratch

  const int prep_total = 3 * CDIM * CDIM + CDIM * CDIM;  // 1048576
  prep_weights<<<(prep_total + 255) / 256, 256, 0, stream>>>(w_qkv, w_proj, wqkvt, wpt);

  qkv_gemm<<<1024 * 24, 32, 0, stream>>>(x, wqkvt, Qh, Kh, Vt);
  attn_kernel<<<BATCH * NHEAD * (NTOK / 16), 32, 0, stream>>>(Qh, Kh, Vt, mask, attn16);
  proj_gemm<<<1024 * 8, 32, 0, stream>>>(attn16, wpt, b_proj, out);
}
